// ShiftedWindowAttention_64381559767408
// MI455X (gfx1250) — matrix operand order repaired and checked by reading, not compiled, not run on hardware
//
#include <hip/hip_runtime.h>
#include <hip/hip_bf16.h>

typedef __attribute__((ext_vector_type(16))) _Float16 v16h;
typedef __attribute__((ext_vector_type(8)))  float    v8f;
typedef __attribute__((ext_vector_type(4)))  _Float16 v4h;
typedef unsigned int u32;
typedef __attribute__((ext_vector_type(4)))  u32      v4u;

#define DIM 128
#define NHEAD 4
#define HD 32
#define WS 7
#define SHIFT 3
#define HH 56
#define NTOK 49
#define NP 64            // padded tokens
#define NEG_INF (-1.0e9f)
__device__ __constant__ float kScale = 0.0883883476483184f; // 128^-0.5

union Frag { v4u u[2]; v16h h; };

// ---- fragment loaders --------------------------------------------------
// A-matrix 16x32 f16: lane<16: row=lane, K=k0..k0+15
//                     lane>=16: row=lane-16, K=k0+16..k0+31
__device__ inline v16h frag_A(const _Float16* base, int row0, int k0, int stride) {
  int lane = threadIdx.x & 31;
  const _Float16* p = base + (row0 + (lane & 15)) * stride + k0 + ((lane >> 4) << 4);
  Frag f;
  f.u[0] = *(const v4u*)p;        // K = k0..k0+7   (+16 for upper half-wave)
  f.u[1] = *(const v4u*)(p + 8);  // K = k0+8..k0+15 (+16 for upper half-wave)
  return f.h;
}
// B-matrix 32x16 f16: lane<16: col N=lane, K=k0..k0+15; lane>=16: K=k0+16..k0+31.
// Source stored "N-major" (rows = N index, contiguous K), i.e. K^T view.
__device__ inline v16h frag_B(const _Float16* base, int col0, int k0, int stride) {
  int lane = threadIdx.x & 31;
  const _Float16* p = base + (col0 + (lane & 15)) * stride + k0 + ((lane >> 4) << 4);
  Frag f;
  f.u[0] = *(const v4u*)p;
  f.u[1] = *(const v4u*)(p + 8);
  return f.h;
}
// Pre-swizzled weight fragment: 32 lanes x 16 f16 contiguous (1024 B per frag)
__device__ inline v16h frag_W(const _Float16* wf, int frag) {
  int lane = threadIdx.x & 31;
  const _Float16* p = wf + frag * 512 + lane * 16;
  Frag f;
  f.u[0] = *(const v4u*)p;
  f.u[1] = *(const v4u*)(p + 8);
  return f.h;
}

__device__ inline v8f wmma16(v16h a, v16h b, v8f c) {
  return __builtin_amdgcn_wmma_f32_16x16x32_f16(false, a, false, b, (short)0, c, false, false);
}

// D layout: c[r] -> row = m0 + r + 8*(lane>=16), col = n0 + (lane&15)
__device__ inline void store_D_f16(_Float16* base, int m0, int n0, int stride, v8f c, float bias) {
  int lane = threadIdx.x & 31;
  int row = m0 + ((lane >> 4) << 3);
  int col = n0 + (lane & 15);
#pragma unroll
  for (int r = 0; r < 8; r++) base[(row + r) * stride + col] = (_Float16)(c[r] + bias);
}
// Transposed store: dst[ch][token], tokens contiguous -> one b128 store per lane
__device__ inline void store_D_T(_Float16* base, int m0, int ch0, v8f c, float bias) {
  int lane = threadIdx.x & 31;
  int tok = m0 + ((lane >> 4) << 3);
  int ch  = ch0 + (lane & 15);
  union { _Float16 h[8]; v4u u; } pk;
#pragma unroll
  for (int r = 0; r < 8; r++) pk.h[r] = (_Float16)(c[r] + bias);
  *(v4u*)(base + ch * 64 + tok) = pk.u;
}

// ---- prep: fp32 weights -> f16 B-fragment-swizzled ---------------------
// qkv_w: K=128 x N=384 (row-major), proj_w: 128x128.
// frag index = nt*4 + kt; element e of lane l: col = nt*16+(l&15),
// k = kt*32 + 16*(l>>4) + e.
__global__ void swin_prep(const float* __restrict__ qkv_w,
                          const float* __restrict__ proj_w,
                          _Float16* __restrict__ wbuf) {
  int tid = blockIdx.x * 256 + threadIdx.x;
  int frag = tid >> 5, lane = tid & 31;
  if (frag >= 128) return;
  const float* W;
  int Ntot, fr;
  if (frag < 96) { W = qkv_w;  Ntot = 384; fr = frag; }
  else           { W = proj_w; Ntot = 128; fr = frag - 96; }
  _Float16* dst = wbuf + frag * 512;
  int nt = fr >> 2, kt = fr & 3;
  int col = nt * 16 + (lane & 15);
  int kb  = kt * 32 + ((lane >> 4) << 4);
  union { _Float16 h[16]; v4u u[2]; } pk;
#pragma unroll
  for (int e = 0; e < 16; e++) pk.h[e] = (_Float16)W[(kb + e) * Ntot + col];
  v4u* o = (v4u*)(dst + lane * 16);
  o[0] = pk.u[0];
  o[1] = pk.u[1];
}

// ---- main: one workgroup per 7x7 window --------------------------------
__global__ __launch_bounds__(256) void swin_main(const float* __restrict__ x,
                                                 const float* __restrict__ qkv_b,
                                                 const float* __restrict__ proj_b,
                                                 const _Float16* __restrict__ wqkv,
                                                 const _Float16* __restrict__ wproj,
                                                 float* __restrict__ out) {
  extern __shared__ char smem[];
  _Float16* s_x  = (_Float16*)smem;            // 64x128 f16 (window x; reused as attn-out)
  _Float16* s_q  = (_Float16*)(smem + 16384);  // 64x128
  _Float16* s_k  = (_Float16*)(smem + 32768);  // 64x128
  _Float16* s_vT = (_Float16*)(smem + 49152);  // 128x64 (channel-major V)
  _Float16* s_s  = (_Float16*)(smem + 65536);  // 64x64 scores -> probs
  int*      s_g  = (int*)(smem + 73728);       // 64 mask-group ids

  const int w  = blockIdx.x;
  const int b  = w >> 6;
  const int wh = (w >> 3) & 7;
  const int ww = w & 7;
  const int wave = threadIdx.x >> 5;
  const int lane = threadIdx.x & 31;

  // mask groups for this window (region labels of the shifted grid)
  if (threadIdx.x < 64) {
    int t = threadIdx.x;
    int tt = t < NTOK ? t : NTOK - 1;
    int i = tt / WS, j = tt % WS;
    int hh = wh * WS + i, wg = ww * WS + j;
    int r = hh < HH - WS ? 0 : (hh < HH - SHIFT ? 1 : 2);
    int c = wg < HH - WS ? 0 : (wg < HH - SHIFT ? 1 : 2);
    s_g[t] = r * 3 + c;
  }

  // phase 0: gather rolled window, fp32 -> f16 into LDS (pad rows 49..63 = 0)
  for (int it = threadIdx.x; it < NP * 32; it += 256) {
    int row = it >> 5, q4 = it & 31;
    float4 v = make_float4(0.f, 0.f, 0.f, 0.f);
    if (row < NTOK) {
      int i = row / WS, j = row % WS;
      int sh = wh * WS + i + SHIFT; if (sh >= HH) sh -= HH;
      int sw = ww * WS + j + SHIFT; if (sw >= HH) sw -= HH;
      v = *(const float4*)(x + (((size_t)b * HH + sh) * HH + sw) * DIM + q4 * 4);
    }
    v4h hv = {(_Float16)v.x, (_Float16)v.y, (_Float16)v.z, (_Float16)v.w};
    *(v4h*)(s_x + row * DIM + q4 * 4) = hv;
  }
  __syncthreads();

  // phase 1: qkv = X(64x128) @ Wqkv(128x384) + b
  // three uniform loops (Q, K, V) so every wave has straight-line stores
  for (int t = wave; t < 32; t += 8) {            // Q: cols 0..127
    int mt = t & 3, nt = t >> 2;
    v8f c = {};
#pragma unroll
    for (int kt = 0; kt < 4; kt++)
      c = wmma16(frag_A(s_x, mt * 16, kt * 32, DIM), frag_W(wqkv, nt * 4 + kt), c);
    store_D_f16(s_q, mt * 16, nt * 16, DIM, c, qkv_b[nt * 16 + (lane & 15)]);
  }
  for (int t = wave; t < 32; t += 8) {            // K: cols 128..255
    int mt = t & 3, nt = t >> 2;
    v8f c = {};
#pragma unroll
    for (int kt = 0; kt < 4; kt++)
      c = wmma16(frag_A(s_x, mt * 16, kt * 32, DIM), frag_W(wqkv, (nt + 8) * 4 + kt), c);
    store_D_f16(s_k, mt * 16, nt * 16, DIM, c, qkv_b[128 + nt * 16 + (lane & 15)]);
  }
  for (int t = wave; t < 32; t += 8) {            // V: cols 256..383, transposed store
    int mt = t & 3, nt = t >> 2;
    v8f c = {};
#pragma unroll
    for (int kt = 0; kt < 4; kt++)
      c = wmma16(frag_A(s_x, mt * 16, kt * 32, DIM), frag_W(wqkv, (nt + 16) * 4 + kt), c);
    store_D_T(s_vT, mt * 16, nt * 16, c, qkv_b[256 + nt * 16 + (lane & 15)]);
  }
  __syncthreads();

  // phase 2: per-head attention (hd=32 -> QK^T tile = single WMMA)
  for (int h = 0; h < NHEAD; h++) {
    for (int t = wave; t < 16; t += 8) {   // 4x4 score tiles
      int mt = t & 3, nt = t >> 2;
      v8f c = {};
      c = wmma16(frag_A(s_q, mt * 16, h * HD, DIM), frag_B(s_k, nt * 16, h * HD, DIM), c);
      store_D_f16(s_s, mt * 16, nt * 16, 64, c, 0.f);
    }
    __syncthreads();

    // masked softmax: 4 lanes per row, 16 f16 scores each, shfl_xor reductions
    {
      int n  = threadIdx.x >> 2;           // row 0..63
      int c0 = (threadIdx.x & 3) * 16;     // column chunk
      int gn = s_g[n];
      union { _Float16 h[16]; v4u u[2]; } in;
      in.u[0] = *(const v4u*)(s_s + n * 64 + c0);
      in.u[1] = *(const v4u*)(s_s + n * 64 + c0 + 8);
      float vals[16];
      float mx = -3.0e38f;
#pragma unroll
      for (int mm = 0; mm < 16; mm++) {
        int m = c0 + mm;
        float vv = -3.0e38f;
        if (m < NTOK)
          vv = (float)in.h[mm] * kScale + ((s_g[m] == gn) ? 0.f : NEG_INF);
        vals[mm] = vv;
        mx = fmaxf(mx, vv);
      }
      mx = fmaxf(mx, __shfl_xor(mx, 1));
      mx = fmaxf(mx, __shfl_xor(mx, 2));
      float sum = 0.f;
#pragma unroll
      for (int mm = 0; mm < 16; mm++) {
        float e = (c0 + mm < NTOK) ? __expf(vals[mm] - mx) : 0.f;
        vals[mm] = e;
        sum += e;
      }
      sum += __shfl_xor(sum, 1);
      sum += __shfl_xor(sum, 2);
      float inv = 1.f / sum;
      union { _Float16 h[16]; v4u u[2]; } pk;
#pragma unroll
      for (int mm = 0; mm < 16; mm++) pk.h[mm] = (_Float16)(vals[mm] * inv);
      *(v4u*)(s_s + n * 64 + c0)     = pk.u[0];
      *(v4u*)(s_s + n * 64 + c0 + 8) = pk.u[1];
    }
    __syncthreads();

    {                                       // O_h = P(64x64) @ V_h ; 8 tiles, 1/wave
      int mt = wave & 3, nt = wave >> 2;    // nt in {0,1}
      v8f c = {};
#pragma unroll
      for (int kt = 0; kt < 2; kt++)
        c = wmma16(frag_A(s_s, mt * 16, kt * 32, 64),
                   frag_B(s_vT, h * HD + nt * 16, kt * 32, 64), c);
      store_D_f16(s_x, mt * 16, h * HD + nt * 16, DIM, c, 0.f); // s_x reused as attn-out
    }
    __syncthreads();
  }

  // phase 3: proj + un-roll scatter to global
  for (int t = wave; t < 32; t += 8) {
    int mt = t & 3, nt = t >> 2;            // nt 0..7
    v8f c = {};
#pragma unroll
    for (int kt = 0; kt < 4; kt++)
      c = wmma16(frag_A(s_x, mt * 16, kt * 32, DIM), frag_W(wproj, nt * 4 + kt), c);
    int col = nt * 16 + (lane & 15);
    float bias = proj_b[col];
    int tok0 = mt * 16 + ((lane >> 4) << 3);
#pragma unroll
    for (int r = 0; r < 8; r++) {
      int tok = tok0 + r;
      if (tok < NTOK) {
        int i = tok / WS, j = tok % WS;
        int dr = wh * WS + i + SHIFT; if (dr >= HH) dr -= HH;
        int dc = ww * WS + j + SHIFT; if (dc >= HH) dc -= HH;
        out[(((size_t)b * HH + dr) * HH + dc) * DIM + col] = c[r] + bias;
      }
    }
  }
}

extern "C" void kernel_launch(void* const* d_in, const int* in_sizes, int n_in,
                              void* d_out, int out_size, void* d_ws, size_t ws_size,
                              hipStream_t stream) {
  const float* x      = (const float*)d_in[0];
  const float* qkv_w  = (const float*)d_in[1];
  const float* qkv_b  = (const float*)d_in[2];
  const float* proj_w = (const float*)d_in[3];
  const float* proj_b = (const float*)d_in[4];
  float* out = (float*)d_out;
  _Float16* wbuf = (_Float16*)d_ws;           // 128 frags * 1024 B = 128 KB

  (void)in_sizes; (void)n_in; (void)out_size; (void)ws_size;

  swin_prep<<<16, 256, 0, stream>>>(qkv_w, proj_w, wbuf);

  const size_t SMEM = 73984;                  // > 64 KB -> raise the cap
  hipFuncSetAttribute((const void*)swin_main,
                      hipFuncAttributeMaxDynamicSharedMemorySize, (int)SMEM);
  swin_main<<<4096, 256, SMEM, stream>>>(x, qkv_b, proj_b, wbuf, wbuf + 96 * 512, out);
}
